// InfoNCELoss_85959475462522
// MI455X (gfx1250) — compile-verified, matrix operand order checked by reading, not run on hardware
//
#include <hip/hip_runtime.h>
#include <hip/hip_bf16.h>

// InfoNCE fused: normalize -> split-bf16 WMMA GEMM (hi*hi + hi*lo + lo*hi)
// -> online exp row-sums -> loss -> mean.  Targets gfx1250 v_wmma_f32_16x16x32_bf16.

typedef __attribute__((ext_vector_type(16))) __bf16 v16bf;
typedef __attribute__((ext_vector_type(8)))  __bf16 v8bf;
typedef __attribute__((ext_vector_type(8)))  float  v8f;

#define NROWS 4096
#define DIM   256
#define ZROWS 8192
#define NTILES 512            // 8192 / 16
#define INV_T 14.2857142857142857f

// ---------------- Phase 1: L2-normalize rows, split into bf16 hi + lo ----------------
__global__ void infonce_normalize_kernel(const float* __restrict__ z1,
                                         const float* __restrict__ z2,
                                         __bf16* __restrict__ zhi,
                                         __bf16* __restrict__ zlo) {
    const int row = blockIdx.x;          // 0..8191
    const int t   = threadIdx.x;         // 0..255 (== DIM)
    const float* src = (row < NROWS) ? (z1 + (size_t)row * DIM)
                                     : (z2 + (size_t)(row - NROWS) * DIM);
    float x = src[t];
    __shared__ float red[DIM];
    red[t] = x * x;
    __syncthreads();
    #pragma unroll
    for (int s = DIM / 2; s > 0; s >>= 1) {
        if (t < s) red[t] += red[t + s];
        __syncthreads();
    }
    const float inv = 1.0f / fmaxf(sqrtf(red[0]), 1e-12f);
    const float y  = x * inv;
    const __bf16 hi = (__bf16)y;
    const __bf16 lo = (__bf16)(y - (float)hi);
    zhi[(size_t)row * DIM + t] = hi;
    zlo[(size_t)row * DIM + t] = lo;
}

// Load one 16x32 bf16 WMMA operand fragment slice for this lane.
// Per ISA layout: lanes 0-15 hold K {0..7, 16..23}, lanes 16-31 hold K {8..15, 24..31}
// (caller bakes the +8 lane offset into `base`); two contiguous 16B chunks.
__device__ __forceinline__ v16bf load_frag(const __bf16* base) {
    union { v16bf v; v8bf h[2]; } u;
    u.h[0] = *(const v8bf*)(base);
    u.h[1] = *(const v8bf*)(base + 16);
    return u.v;
}

// ---------------- Phase 2: WMMA GEMM + fused exp row-sum / pos / col-n capture ----------------
__global__ __launch_bounds__(256, 1)
void infonce_gemm_kernel(const __bf16* __restrict__ zhi,
                         const __bf16* __restrict__ zlo,
                         float* __restrict__ lossRow) {
    const int row0    = blockIdx.x * 16;        // M tile base (rows of z1n)
    const int lane    = threadIdx.x & 31;
    const int wave    = threadIdx.x >> 5;       // 0..7
    const int laneMod = lane & 15;
    const int laneHi  = lane >> 4;              // 0 or 1
    const int kLane   = laneHi ? 8 : 0;

    __shared__ float redS[16];   // per-row exp sums (cross-wave)
    __shared__ float sposS[16];  // s[i, n+i] (scaled)
    __shared__ float expnS[16];  // exp(s[i, n])
    if (threadIdx.x < 16) redS[threadIdx.x] = 0.0f;
    __syncthreads();

    // A fragments resident: 16 rows x 256 K, hi + lo  (8 K-steps of 32)
    v16bf a_hi[8], a_lo[8];
    {
        const size_t arow = (size_t)(row0 + laneMod) * DIM;
        #pragma unroll
        for (int k = 0; k < 8; ++k) {
            const size_t off = arow + 32 * k + kLane;
            a_hi[k] = load_frag(zhi + off);
            a_lo[k] = load_frag(zlo + off);
        }
    }

    float accSum[8];
    #pragma unroll
    for (int r = 0; r < 8; ++r) accSum[r] = 0.0f;

    const int jposTile = 256 + blockIdx.x;      // tile holding columns 4096+row0..+15
    const int jnTile   = 256;                   // tile holding column n = 4096

    for (int jt = wave; jt < NTILES; jt += 8) {
        const size_t brow = (size_t)(jt * 16 + laneMod) * DIM;
        // prefetch next tile's B row into caches (global_prefetch_b8)
        if (jt + 8 < NTILES)
            __builtin_prefetch(zhi + brow + (size_t)128 * DIM, 0, 1);

        v8f acc = {};
        #pragma unroll
        for (int k = 0; k < 8; ++k) {
            const size_t off = brow + 32 * k + kLane;
            const v16bf b_hi = load_frag(zhi + off);
            const v16bf b_lo = load_frag(zlo + off);
            acc = __builtin_amdgcn_wmma_f32_16x16x32_bf16(false, a_hi[k], false, b_hi,
                                                          (short)0, acc, false, false);
            acc = __builtin_amdgcn_wmma_f32_16x16x32_bf16(false, a_hi[k], false, b_lo,
                                                          (short)0, acc, false, false);
            acc = __builtin_amdgcn_wmma_f32_16x16x32_bf16(false, a_lo[k], false, b_hi,
                                                          (short)0, acc, false, false);
        }
        // C/D layout: lane gives N=lane%16; vgpr r gives M = r + 8*(lane/16)
        #pragma unroll
        for (int r = 0; r < 8; ++r) {
            const float s = acc[r] * INV_T;
            const float e = __expf(s);
            accSum[r] += e;
            const int M = r + 8 * laneHi;
            if (jt == jposTile && laneMod == M) sposS[M] = s;   // diagonal positive
            if (jt == jnTile   && laneMod == 0) expnS[M] = e;   // excluded column n
        }
    }

    // reduce exp sums across the 16 lanes of each half (wave32: xor 1,2,4,8 stays in-half)
    #pragma unroll
    for (int r = 0; r < 8; ++r) {
        float v = accSum[r];
        v += __shfl_xor(v, 1, 32);
        v += __shfl_xor(v, 2, 32);
        v += __shfl_xor(v, 4, 32);
        v += __shfl_xor(v, 8, 32);
        accSum[r] = v;
    }
    if (laneMod == 0) {
        #pragma unroll
        for (int r = 0; r < 8; ++r)
            atomicAdd(&redS[r + 8 * laneHi], accSum[r]);
    }
    __syncthreads();

    if (threadIdx.x < 16) {
        const int M = threadIdx.x;
        const float sumNeg = redS[M] - expnS[M];     // total minus column n (ref semantics)
        lossRow[row0 + M] = logf(sumNeg) - sposS[M]; // -log(exp(spos)/sumNeg)
    }
}

// ---------------- Phase 3: deterministic mean over 4096 per-row losses ----------------
__global__ void infonce_reduce_kernel(const float* __restrict__ lossRow,
                                      float* __restrict__ out) {
    __shared__ float red[256];
    const int t = threadIdx.x;
    float s = 0.0f;
    for (int i = t; i < NROWS; i += 256) s += lossRow[i];
    red[t] = s;
    __syncthreads();
    #pragma unroll
    for (int k = 128; k > 0; k >>= 1) {
        if (t < k) red[t] += red[t + k];
        __syncthreads();
    }
    if (t == 0) out[0] = red[0] * (1.0f / NROWS);
}

extern "C" void kernel_launch(void* const* d_in, const int* in_sizes, int n_in,
                              void* d_out, int out_size, void* d_ws, size_t ws_size,
                              hipStream_t stream) {
    (void)in_sizes; (void)n_in; (void)out_size; (void)ws_size;
    const float* z1 = (const float*)d_in[0];   // [4096,256] f32
    const float* z2 = (const float*)d_in[1];   // [4096,256] f32

    char* ws = (char*)d_ws;
    const size_t zbytes = (size_t)ZROWS * DIM * sizeof(__bf16);   // 4 MB each
    __bf16* zhi    = (__bf16*)ws;
    __bf16* zlo    = (__bf16*)(ws + zbytes);
    float*  lossRow = (float*)(ws + 2 * zbytes);                  // 16 KB

    infonce_normalize_kernel<<<ZROWS, DIM, 0, stream>>>(z1, z2, zhi, zlo);
    infonce_gemm_kernel<<<NROWS / 16, 256, 0, stream>>>(zhi, zlo, lossRow);
    infonce_reduce_kernel<<<1, 256, 0, stream>>>(lossRow, (float*)d_out);
}
